// OptimizedLinearAttention_79353815761051
// MI455X (gfx1250) — compile-verified, matrix operand order checked by reading, not run on hardware
//
#include <hip/hip_runtime.h>

typedef __bf16 bf16;
typedef __attribute__((ext_vector_type(16))) __bf16 v16bf;
typedef __attribute__((ext_vector_type(8)))  float  v8f;

#define NHEADS   8
#define DHEAD    32
#define NB       8          // batch
#define NC       256        // channels / dim
#define NN       16384      // H*W
#define QKV_ROWS 768
#define SCALE_Q  0.17677669529663687f
#define EPS_LN   1e-5f
#define QROW     272        // 256 + 16 pad (row stride stays 32B-aligned)
#define CROW     144        // 128 + 16 pad for context tiles (288B rows)

// ---------------- kernel 1: per-pixel channel mean/rstd (LN over C) -------
__global__ void ln_stats_kernel(const float* __restrict__ x,
                                float* __restrict__ mean, float* __restrict__ rstd) {
  int idx = blockIdx.x * blockDim.x + threadIdx.x;     // b*NN + n
  int b = idx >> 14, n = idx & (NN - 1);
  const float* xp = x + ((long)b << 22) + n;           // b * C * NN
  float s = 0.f, s2 = 0.f;
#pragma unroll 4
  for (int c = 0; c < NC; ++c) {
    float v = xp[(long)c << 14];
    s += v; s2 += v * v;
  }
  float mu  = s * (1.0f / NC);
  float var = s2 * (1.0f / NC) - mu * mu;
  mean[idx] = mu;
  rstd[idx] = rsqrtf(var + EPS_LN);
}

// ---------------- kernel 2: qkv = W_qkv @ norm(x), bf16 WMMA --------------
// block: 256 threads (8 waves). tile: M=128 (wave=16 rows), N=128, K=256.
__global__ void qkv_gemm_kernel(const float* __restrict__ x,
                                const float* __restrict__ mean, const float* __restrict__ rstd,
                                const float* __restrict__ g,    const float* __restrict__ bch,
                                const float* __restrict__ w_qkv,
                                bf16* __restrict__ qkv) {
  extern __shared__ __align__(128) char smem[];
  bf16* xt = (bf16*)smem;                               // [128][QROW] = norm-x tile, [n][c]
  int t  = threadIdx.x;
  int n0 = blockIdx.x * 128;
  int m0 = blockIdx.y * 128;
  int z  = blockIdx.z;
  { // stage normalized x tile, transposed to [n][c] for WMMA B-operand
    int nl = t & 127;
    int ch = (t >> 7) * 128;
    int gn = n0 + nl;
    float mu = mean[(z << 14) + gn];
    float rs = rstd[(z << 14) + gn];
    const float* xp = x + ((long)z << 22) + gn;
#pragma unroll 4
    for (int i = 0; i < 128; ++i) {
      int c = ch + i;
      float xn = (xp[(long)c << 14] - mu) * rs * g[c] + bch[c];
      xt[nl * QROW + c] = (bf16)xn;
    }
  }
  __syncthreads();

  int wave = t >> 5, lane = t & 31;
  int mrow = m0 + wave * 16 + (lane & 15);
  int kh   = (lane >> 4) * 16;
  v8f acc[8];
#pragma unroll
  for (int i = 0; i < 8; ++i) acc[i] = (v8f){0.f,0.f,0.f,0.f,0.f,0.f,0.f,0.f};

#pragma unroll
  for (int kk = 0; kk < 256; kk += 32) {
    const float* ap = w_qkv + mrow * 256 + kk + kh;     // L2-resident weights
    v16bf A;
#pragma unroll
    for (int j = 0; j < 16; ++j) A[j] = (bf16)ap[j];
#pragma unroll
    for (int nb = 0; nb < 8; ++nb) {
      v16bf Bm = *(const v16bf*)(xt + (nb * 16 + (lane & 15)) * QROW + kk + kh);
      acc[nb] = __builtin_amdgcn_wmma_f32_16x16x32_bf16(false, A, false, Bm,
                                                        (short)0, acc[nb], false, false);
    }
  }
  float sc = (m0 < 256) ? SCALE_Q : 1.0f;               // pre-scale q rows
  int mbase = m0 + wave * 16 + ((lane >> 4) << 3);
  bf16* outp = qkv + ((long)z * QKV_ROWS << 14);
#pragma unroll
  for (int nb = 0; nb < 8; ++nb) {
    int n = n0 + nb * 16 + (lane & 15);
#pragma unroll
    for (int r = 0; r < 8; ++r)
      outp[((long)(mbase + r) << 14) + n] = (bf16)(acc[nb][r] * sc);
  }
}

// ---------------- kernel 3: per-row (b,h,c) max & sumexp of k -------------
__global__ void softmax_stats_kernel(const bf16* __restrict__ qkv,
                                     float* __restrict__ rowmax, float* __restrict__ rowsum) {
  int wave = threadIdx.x >> 5, lane = threadIdx.x & 31;
  int row = blockIdx.x * 8 + wave;                      // 0..2047
  int b = row >> 8, j = row & 255;
  const bf16* kp = qkv + (((long)b * QKV_ROWS + 256 + j) << 14);
  float m = -3.4e38f;
  for (int n = lane; n < NN; n += 32) m = fmaxf(m, (float)kp[n]);
#pragma unroll
  for (int off = 16; off > 0; off >>= 1) m = fmaxf(m, __shfl_xor(m, off, 32));
  float s = 0.f;
  for (int n = lane; n < NN; n += 32) s += __expf((float)kp[n] - m);
#pragma unroll
  for (int off = 16; off > 0; off >>= 1) s += __shfl_xor(s, off, 32);
  if (lane == 0) { rowmax[row] = m; rowsum[row] = s; }
}

// ---------------- kernel 4: k <- exp(k - m)/Z  (in place, bf16) -----------
__global__ void softmax_apply_kernel(bf16* __restrict__ qkv,
                                     const float* __restrict__ rowmax,
                                     const float* __restrict__ rowsum) {
  long total = (long)NB * 256 * NN;
  long step  = (long)gridDim.x * blockDim.x;
  for (long idx = (long)blockIdx.x * blockDim.x + threadIdx.x; idx < total; idx += step) {
    int row = (int)(idx >> 14);
    int n   = (int)(idx & (NN - 1));
    int b = row >> 8, j = row & 255;
    bf16* kp = qkv + (((long)b * QKV_ROWS + 256 + j) << 14);
    kp[n] = (bf16)(__expf((float)kp[n] - rowmax[row]) / rowsum[row]);
  }
}

// ---------------- kernel 5: partial context via async-LDS + WMMA ----------
// grid (64, 16): (b*8+h, n-chunk of 1024). 128 threads = 4 waves.
// ek/v tiles (32 x 128 bf16) double-buffered in LDS via
// GLOBAL_LOAD_ASYNC_TO_LDS_B128 (ASYNCcnt pipeline), partials accumulated
// into ctx with global f32 atomics.
__global__ void context_partial_kernel(const bf16* __restrict__ qkv,
                                       float* __restrict__ ctx) {
  __shared__ __align__(128) bf16 ekb[2][32 * CROW];
  __shared__ __align__(128) bf16 vb [2][32 * CROW];
  int bh = blockIdx.x;                                  // 0..63
  int chunk = blockIdx.y;                               // 0..15
  int b = bh >> 3, h = bh & 7;
  int t = threadIdx.x;
  const bf16* ekbase = qkv + (((long)b * QKV_ROWS + 256 + h * 32) << 14) + chunk * 1024;
  const bf16* vbase  = qkv + (((long)b * QKV_ROWS + 512 + h * 32) << 14) + chunk * 1024;

  auto stage = [&](int bi, int co) {                    // stage 2 x 8KB tiles
#pragma unroll
    for (int i = 0; i < 4; ++i) {
      int linear = t + i * 128;                         // 0..511
      int row = linear >> 4;                            // 0..31
      int seg = linear & 15;                            // 16B segments per row
      unsigned lde = (unsigned)(unsigned long long)&ekb[bi][row * CROW + seg * 8];
      unsigned long long gae =
          (unsigned long long)(ekbase + ((long)row << 14) + co + seg * 8);
      asm volatile("global_load_async_to_lds_b128 %0, %1, off"
                   :: "v"(lde), "v"(gae) : "memory");
      unsigned ldv = (unsigned)(unsigned long long)&vb[bi][row * CROW + seg * 8];
      unsigned long long gav =
          (unsigned long long)(vbase + ((long)row << 14) + co + seg * 8);
      asm volatile("global_load_async_to_lds_b128 %0, %1, off"
                   :: "v"(ldv), "v"(gav) : "memory");
    }
  };

  int wave = t >> 5, lane = t & 31;
  int ci = wave >> 1, di = wave & 1;
  int arow = ci * 16 + (lane & 15);
  int brow = di * 16 + (lane & 15);
  int kh = (lane >> 4) * 16;
  v8f acc = (v8f){0.f,0.f,0.f,0.f,0.f,0.f,0.f,0.f};

  stage(0, 0);
  int cur = 0;
  for (int st = 0; st < 8; ++st) {
    asm volatile("s_wait_asynccnt 0" ::: "memory");
    __syncthreads();                                    // tile[cur] visible to all
    if (st + 1 < 8) stage(cur ^ 1, (st + 1) * 128);     // prefetch next tile
#pragma unroll
    for (int kk = 0; kk < 128; kk += 32) {
      v16bf A  = *(const v16bf*)(&ekb[cur][arow * CROW + kk + kh]);
      v16bf Bm = *(const v16bf*)(&vb [cur][brow * CROW + kk + kh]);
      acc = __builtin_amdgcn_wmma_f32_16x16x32_bf16(false, A, false, Bm,
                                                    (short)0, acc, false, false);
    }
    __syncthreads();                                    // done reading tile[cur]
    cur ^= 1;
  }
  int mrel = ci * 16 + ((lane >> 4) << 3);
  float* cp = ctx + (long)bh * 1024;                    // [c][d] row-major
#pragma unroll
  for (int r = 0; r < 8; ++r)
    atomicAdd(&cp[(mrel + r) * 32 + di * 16 + (lane & 15)], acc[r]);
}

// ---------------- kernel 6: fused attn-out + W_out GEMM + residual --------
// block: 256 threads, tile N=128. GEMM1: per-head ctx^T(32x32) @ q(32x128).
// GEMM2: W_out(256x256) @ attn(256x128) + b_out + xn.
__global__ void out_kernel(const float* __restrict__ x,
                           const float* __restrict__ mean, const float* __restrict__ rstd,
                           const float* __restrict__ g,    const float* __restrict__ bch,
                           const bf16* __restrict__ qkv,   const float* __restrict__ ctx,
                           const float* __restrict__ w_out,const float* __restrict__ b_out,
                           float* __restrict__ out) {
  extern __shared__ __align__(128) char smem[];
  bf16* qt    = (bf16*)smem;                            // [128][QROW]  q tile [n][c]
  bf16* attnT = (bf16*)(smem + 128 * QROW * 2);         // [128][QROW]  attn [n][hd]
  bf16* ctxT  = (bf16*)(smem + 2 * 128 * QROW * 2);     // [8][32][32]  ctx^T [h][d][c]
  int t  = threadIdx.x;
  int n0 = blockIdx.x * 128;
  int z  = blockIdx.z;
  { // stage q tile (pre-scaled bf16) transposed to [n][c]; 4B coalesced loads
    int nl2 = (t & 63) * 2;
    int ch  = (t >> 6) * 64;
    const bf16* qp = qkv + ((long)z * QKV_ROWS << 14) + n0 + nl2;
#pragma unroll 4
    for (int i = 0; i < 64; ++i) {
      int c = ch + i;
      const bf16* p = qp + ((long)c << 14);
      bf16 lo = p[0], hi = p[1];                        // adjacent -> one b32 load
      qt[nl2 * QROW + c]       = lo;
      qt[(nl2 + 1) * QROW + c] = hi;
    }
  }
#pragma unroll
  for (int i = 0; i < 32; ++i) {                        // stage transposed contexts
    int lin = t + i * 256;                              // 0..8191
    int h = lin >> 10, cc = (lin >> 5) & 31, dd = lin & 31;
    ctxT[h * 1024 + dd * 32 + cc] = (bf16)ctx[((long)z << 13) + lin];
  }
  __syncthreads();

  int wave = t >> 5, lane = t & 31;
  int kh = (lane >> 4) * 16;
  { // GEMM1: one head per wave, K = 32 (single WMMA per output tile)
    int h = wave;
#pragma unroll
    for (int msub = 0; msub < 2; ++msub) {
      v16bf A = *(const v16bf*)(ctxT + h * 1024 + (msub * 16 + (lane & 15)) * 32 + kh);
#pragma unroll
      for (int nb = 0; nb < 8; ++nb) {
        v16bf Bm = *(const v16bf*)(qt + (nb * 16 + (lane & 15)) * QROW + h * 32 + kh);
        v8f acc = (v8f){0.f,0.f,0.f,0.f,0.f,0.f,0.f,0.f};
        acc = __builtin_amdgcn_wmma_f32_16x16x32_bf16(false, A, false, Bm,
                                                      (short)0, acc, false, false);
        int n  = nb * 16 + (lane & 15);
        int hd = h * 32 + msub * 16 + ((lane >> 4) << 3);
#pragma unroll
        for (int r = 0; r < 8; ++r)
          attnT[n * QROW + hd + r] = (bf16)acc[r];
      }
    }
  }
  __syncthreads();

  // GEMM2: wave owns 32 output rows (o = wave*32 .. +31)
  v8f acc2[2][8];
#pragma unroll
  for (int m = 0; m < 2; ++m)
#pragma unroll
    for (int nb = 0; nb < 8; ++nb) acc2[m][nb] = (v8f){0.f,0.f,0.f,0.f,0.f,0.f,0.f,0.f};

#pragma unroll
  for (int kk = 0; kk < 256; kk += 32) {
    const float* a0 = w_out + (wave * 32 + (lane & 15)) * 256 + kk + kh;
    const float* a1 = a0 + 16 * 256;
    v16bf A0, A1;
#pragma unroll
    for (int j = 0; j < 16; ++j) { A0[j] = (bf16)a0[j]; A1[j] = (bf16)a1[j]; }
#pragma unroll
    for (int nb = 0; nb < 8; ++nb) {
      v16bf Bm = *(const v16bf*)(attnT + (nb * 16 + (lane & 15)) * QROW + kk + kh);
      acc2[0][nb] = __builtin_amdgcn_wmma_f32_16x16x32_bf16(false, A0, false, Bm,
                                                            (short)0, acc2[0][nb], false, false);
      acc2[1][nb] = __builtin_amdgcn_wmma_f32_16x16x32_bf16(false, A1, false, Bm,
                                                            (short)0, acc2[1][nb], false, false);
    }
  }
  // epilogue: + b_out + xn (recomputed from x and cached stats)
#pragma unroll
  for (int nb = 0; nb < 8; ++nb) {
    int n = n0 + nb * 16 + (lane & 15);
    float mu = mean[(z << 14) + n];
    float rs = rstd[(z << 14) + n];
#pragma unroll
    for (int msub = 0; msub < 2; ++msub) {
      int obase = wave * 32 + msub * 16 + ((lane >> 4) << 3);
#pragma unroll
      for (int r = 0; r < 8; ++r) {
        int o = obase + r;
        long off = ((long)((z << 8) + o) << 14) + n;
        float xn = (x[off] - mu) * rs * g[o] + bch[o];
        out[off] = acc2[msub][nb][r] + b_out[o] + xn;
      }
    }
  }
}

extern "C" void kernel_launch(void* const* d_in, const int* in_sizes, int n_in,
                              void* d_out, int out_size, void* d_ws, size_t ws_size,
                              hipStream_t stream) {
  const float* x     = (const float*)d_in[0];
  const float* g     = (const float*)d_in[1];
  const float* b     = (const float*)d_in[2];
  const float* w_qkv = (const float*)d_in[3];
  const float* w_out = (const float*)d_in[4];
  const float* b_out = (const float*)d_in[5];
  float* out = (float*)d_out;

  float* ws     = (float*)d_ws;
  float* mean   = ws;                                   // B*N
  float* rstd   = mean + (long)NB * NN;                 // B*N
  float* rowmax = rstd + (long)NB * NN;                 // 2048
  float* rowsum = rowmax + 2048;                        // 2048
  float* ctx    = rowsum + 2048;                        // 8*8*32*32
  bf16*  qkv    = (bf16*)(ctx + (long)NB * NHEADS * 32 * 32); // 8*768*16384 bf16

  hipMemsetAsync(ctx, 0, (size_t)NB * NHEADS * 32 * 32 * sizeof(float), stream);

  ln_stats_kernel<<<dim3((NB * NN) / 256), 256, 0, stream>>>(x, mean, rstd);

  size_t lds_b = (size_t)128 * QROW * 2;                // 68 KB
  hipFuncSetAttribute((const void*)qkv_gemm_kernel,
                      hipFuncAttributeMaxDynamicSharedMemorySize, (int)lds_b);
  qkv_gemm_kernel<<<dim3(NN / 128, 6, NB), 256, lds_b, stream>>>(
      x, mean, rstd, g, b, w_qkv, qkv);

  softmax_stats_kernel<<<dim3(256), 256, 0, stream>>>(qkv, rowmax, rowsum);
  softmax_apply_kernel<<<dim3(16384), 256, 0, stream>>>(qkv, rowmax, rowsum);
  context_partial_kernel<<<dim3(64, 16), 128, 0, stream>>>(qkv, ctx);

  size_t lds_e = (size_t)2 * 128 * QROW * 2 + 8 * 32 * 32 * 2; // 152 KB
  hipFuncSetAttribute((const void*)out_kernel,
                      hipFuncAttributeMaxDynamicSharedMemorySize, (int)lds_e);
  out_kernel<<<dim3(NN / 128, 1, NB), 256, lds_e, stream>>>(
      x, mean, rstd, g, b, qkv, ctx, w_out, b_out, out);
}
